// Hyperkernel_14027363188950
// MI455X (gfx1250) — matrix operand order repaired and checked by reading, not compiled, not run on hardware
//
#include <hip/hip_runtime.h>

typedef __attribute__((ext_vector_type(16))) __bf16 v16bf;
typedef __attribute__((ext_vector_type(8)))  __bf16 v8bf;
typedef __attribute__((ext_vector_type(8)))  float  v8f;
typedef __attribute__((ext_vector_type(4)))  float  v4f;

#define BB 8
#define CC 16
#define EE 256
#define HH 256
#define WW 256
#define KPAD 160          // 10 taps (9 real + 1 zero pad) * 16 channels
#define BN 64             // w-tile per block (E fully covered per block)
#define NCOL 72           // slab columns: fragment reads cols 3..68
#define CSTR 24           // bf16 per (row,col) slot; 48B -> banks tile, 16B aligned

// ---------------- prep: gather + pad + bf16-convert weights ----------------
// Aglob[b][e][k], k = t*16 + c, t in 0..9 (t==9 zero pad), c in 0..15
__global__ __launch_bounds__(256) void
gather_weights_bf16(const int* __restrict__ indices,
                    const float* __restrict__ emb,
                    __bf16* __restrict__ Aglob)
{
    const int id = blockIdx.x * 256 + threadIdx.x;   // 20480 tasks: (b, e, t)
    const int b  = id / 2560;
    const int r1 = id % 2560;
    const int e  = r1 / 10;
    const int t  = r1 % 10;

    v8bf lo, hi;
#pragma unroll
    for (int c = 0; c < CC; ++c) {
        float v = 0.f;
        if (t < 9)
            v = emb[(size_t)indices[b * CC + c] * (EE * 9) + (size_t)e * 9 + t];
        __bf16 bv = (__bf16)v;
        if (c < 8) lo[c] = bv; else hi[c - 8] = bv;
    }
    __bf16* dst = Aglob + ((size_t)(b * EE + e) * KPAD + t * 16);
    *(v8bf*)(dst)     = lo;                          // 32B-aligned
    *(v8bf*)(dst + 8) = hi;
}

__device__ __forceinline__ v16bf join16(v8bf lo, v8bf hi) {
    return __builtin_shufflevector(lo, hi,
               0,1,2,3,4,5,6,7, 8,9,10,11,12,13,14,15);
}

// ---------------- main: slab-staged implicit-im2col WMMA GEMM ----------------
// One block covers ALL of E (4 M-tiles of 64) for a 64-wide w strip of one h row.
__global__ __launch_bounds__(256) void
dyn_conv_wmma(const float* __restrict__ x,
              const __bf16* __restrict__ Aglob,
              const float* __restrict__ bias,
              float* __restrict__ out)
{
    // slabT[row 0..3][col 0..71][c 0..15]; row 3 = zeros (t==9 pad)
    __shared__ __attribute__((aligned(16))) __bf16 slabT[4 * NCOL * CSTR];

    const int tid = threadIdx.x;
    const int b   = blockIdx.z;
    const int tw  = blockIdx.x & 3;       // W / BN = 4 tiles per row
    const int h   = blockIdx.x >> 2;
    const int w0  = tw * BN;

    // zero pad row 3 (only slots [0..15] of each col are ever read)
    if (tid < NCOL * 2) {
        const int col = tid >> 1, hv = tid & 1;
        v8bf z = {};
        *(v8bf*)&slabT[(3 * NCOL + col) * CSTR + hv * 8] = z;
    }
    // interior columns: aligned float4 loads; tasks (c, row, m), m = 4-col chunk
    for (int t3 = tid; t3 < CC * 3 * 16; t3 += 256) {   // 768/256 = 3, uniform
        const int c  = t3 / 48;
        const int rr = (t3 / 16) % 3;
        const int m  = t3 & 15;
        const int hh = h + rr - 1;
        v4f v = {};
        if (hh >= 0 && hh < HH)
            v = *(const v4f*)&x[(((size_t)b * CC + c) * HH + hh) * WW + w0 + 4 * m];
        const int col = 4 * m + 4;                      // slab col = xcol - w0 + 4
#pragma unroll
        for (int j = 0; j < 4; ++j)
            slabT[(rr * NCOL + col + j) * CSTR + c] = (__bf16)v[j];
    }
    // edge columns: xcol = w0-1 (col 3) and xcol = w0+64 (col 68)
    if (tid < CC * 3 * 2) {
        const int c    = tid / 6;
        const int rr   = (tid / 2) % 3;
        const int side = tid & 1;
        const int hh   = h + rr - 1;
        const int xcol = side ? (w0 + BN) : (w0 - 1);
        const int col  = side ? 68 : 3;
        float v = 0.f;
        if (hh >= 0 && hh < HH && xcol >= 0 && xcol < WW)
            v = x[(((size_t)b * CC + c) * HH + hh) * WW + xcol];
        slabT[(rr * NCOL + col) * CSTR + c] = (__bf16)v;
    }
    __syncthreads();

    const int lane = tid & 31;
    const int half = lane >> 4;           // lane group 0-15 / 16-31
    const int lrow = lane & 15;
    const int wave = tid >> 5;
    const int wm   = wave & 3;            // 4 sub-M-tiles of 16 within each 64
    const int wn   = wave >> 2;           // 2 N-pairs
    const int em   = wm * 16;
    const int n0   = wn * 32;
    const int n1   = n0 + 16;

    // A rows for the 4 M-tiles (64 apart in E), contiguous K per row
    const __bf16* arow = Aglob + (size_t)(b * EE + em + lrow) * KPAD;

    v8f acc0[4] = {{}, {}, {}, {}};
    v8f acc1[4] = {{}, {}, {}, {}};

#pragma unroll
    for (int kt = 0; kt < KPAD / 32; ++kt) {
        const int k0 = kt * 32;
        // B fragment: K = 16*t + c with t = 2*kt + half fixed per lane-half;
        // elements are the 16 channels, contiguous in slabT -> 2x ds_load_b128.
        // Shared across all 4 M-tiles (operand reuse).
        const int t  = 2 * kt + half;
        const int kh = t / 3;              // t==9 -> row 3 (zeros)
        const int kw = t - 3 * kh;
        const __bf16* s0 = &slabT[(kh * NCOL + (n0 + lrow + 3 + kw)) * CSTR];
        const __bf16* s1 = s0 + 16 * CSTR;                 // n1 = n0 + 16
        v16bf bf0 = join16(*(const v8bf*)(s0), *(const v8bf*)(s0 + 8));
        v16bf bf1 = join16(*(const v8bf*)(s1), *(const v8bf*)(s1 + 8));

#pragma unroll
        for (int m = 0; m < 4; ++m) {
            const __bf16* ar = arow + (size_t)(m * 64) * KPAD;
            // A 16x32 bf16: K runs [k0+half*16, +16) contiguous (matches B)
            v16bf a = join16(*(const v8bf*)(ar + k0 + half * 16),
                             *(const v8bf*)(ar + k0 + half * 16 + 8));
            acc0[m] = __builtin_amdgcn_wmma_f32_16x16x32_bf16(
                          false, a, false, bf0, (short)0, acc0[m], false, false);
            acc1[m] = __builtin_amdgcn_wmma_f32_16x16x32_bf16(
                          false, a, false, bf1, (short)0, acc1[m], false, false);
        }
    }

    // store D: C/D layout M = half*8 + vgpr, N = lrow
#pragma unroll
    for (int m = 0; m < 4; ++m) {
        const int ebase = m * 64 + em + half * 8;          // multiple of 8
        const v4f bl = *(const v4f*)&bias[ebase];
        const v4f bh = *(const v4f*)&bias[ebase + 4];
        float* obase = out + (((size_t)(b * EE + ebase)) * HH + h) * WW + w0;
#pragma unroll
        for (int g = 0; g < 8; ++g) {
            const float bv = (g < 4) ? bl[g] : bh[g - 4];
            float* op = obase + (size_t)g * (HH * WW);     // immediate-offset friendly
            op[n0 + lrow] = acc0[m][g] + bv;
            op[n1 + lrow] = acc1[m][g] + bv;
        }
    }
}

extern "C" void kernel_launch(void* const* d_in, const int* in_sizes, int n_in,
                              void* d_out, int out_size, void* d_ws, size_t ws_size,
                              hipStream_t stream) {
    const float* x    = (const float*)d_in[0];
    const int*   idx  = (const int*)d_in[1];
    const float* emb  = (const float*)d_in[2];
    const float* bias = (const float*)d_in[3];
    float* out = (float*)d_out;
    __bf16* Aglob = (__bf16*)d_ws;        // 8*256*160*2 = 655,360 B

    hipLaunchKernelGGL(gather_weights_bf16, dim3(80), dim3(256), 0, stream,
                       idx, emb, Aglob);

    dim3 grid(HH * (WW / BN), 1, BB);     // (1024, 1, 8): all of E per block
    hipLaunchKernelGGL(dyn_conv_wmma, grid, dim3(256), 0, stream,
                       x, Aglob, bias, out);
}
